// GenericVisionSelfAttentionKernel_57672820850771
// MI455X (gfx1250) — matrix operand order repaired and checked by reading, not compiled, not run on hardware
//
#include <hip/hip_runtime.h>
#include <hip/hip_bf16.h>

typedef __bf16 bf16_t;
typedef __attribute__((ext_vector_type(16))) __bf16 v16bf;
typedef __attribute__((ext_vector_type(8)))  __bf16 v8bf;
typedef __attribute__((ext_vector_type(4)))  __bf16 v4bf;
typedef __attribute__((ext_vector_type(8)))  float  v8f;
typedef __attribute__((ext_vector_type(4)))  unsigned int u32x4;
typedef __attribute__((ext_vector_type(8)))  int i32x8;
typedef __attribute__((ext_vector_type(4)))  int i32x4;

#define B_  8
#define S_  1024
#define D_  1024
#define H_  16
#define HD_ 64

static __device__ __forceinline__ v16bf cat8(v8bf lo, v8bf hi) {
  return __builtin_shufflevector(lo, hi, 0,1,2,3,4,5,6,7,8,9,10,11,12,13,14,15);
}

// LDS byte offset of a shared-memory pointer (flat LDS aperture: addr[31:0]).
static __device__ __forceinline__ unsigned lds_addr32(const void* p) {
  return (unsigned)(unsigned long long)p;
}

// ---------------------------------------------------------------------------
// TDM: 2-D bf16 tile load, global -> LDS (D# groups per ISA 08_async_tensor §8).
// Optional LDS row padding: pad_interval_code / pad_amount_code per D# group1.
// ---------------------------------------------------------------------------
static __device__ __forceinline__ void tdm_load_2d_bf16(
    unsigned lds_byte_addr, const bf16_t* gptr,
    unsigned tile_d0, unsigned tile_d1,
    unsigned tensor_d0, unsigned tensor_d1, unsigned stride_d0_elems,
    unsigned pad_interval_code, unsigned pad_amount_code, bool pad_en) {
  const unsigned long long ga = (unsigned long long)gptr;
  u32x4 g0;
  g0[0] = 1u;                                              // count=1, user mode
  g0[1] = lds_byte_addr;                                   // lds_addr
  g0[2] = (unsigned)(ga & 0xffffffffull);                  // global_addr[31:0]
  g0[3] = (unsigned)((ga >> 32) & 0x01ffffffull) | (2u << 30);  // [56:32] | type=2
  i32x8 g1;
  unsigned dw0 = (1u << 16);                               // data_size=1 (2 bytes)
  if (pad_en)
    dw0 |= (1u << 20) | (pad_interval_code << 22) | (pad_amount_code << 25);
  g1[0] = (int)dw0;                                        // wg_mask=0 (no cluster)
  g1[1] = (int)((tensor_d0 & 0xffffu) << 16);              // tensor_dim0[15:0]
  g1[2] = (int)(((tensor_d0 >> 16) & 0xffffu) |
                ((tensor_d1 & 0xffffu) << 16));            // dim0[31:16]|dim1[15:0]
  g1[3] = (int)(((tensor_d1 >> 16) & 0xffffu) |
                ((tile_d0 & 0xffffu) << 16));              // dim1[31:16]|tile0
  g1[4] = (int)(tile_d1 & 0xffffu);                        // tile_dim1, tile_dim2=0
  g1[5] = (int)stride_d0_elems;                            // tensor_dim0_stride lo
  g1[6] = 0;                                               // stride hi, dim1_stride lo
  g1[7] = 0;
  const i32x4 z4 = {0, 0, 0, 0};
#if defined(__clang_major__) && (__clang_major__ >= 23)
  const i32x8 z8 = {0, 0, 0, 0, 0, 0, 0, 0};
  __builtin_amdgcn_tensor_load_to_lds(g0, g1, z4, z4, z8, 0);
#else
  __builtin_amdgcn_tensor_load_to_lds(g0, g1, z4, z4, 0);
#endif
}

// ---------------------------------------------------------------------------
// fp32 -> bf16 conversion (softmax 1/sqrt(HD) folded into Wq)
// ---------------------------------------------------------------------------
__global__ void cvt_f32_to_bf16(const float* __restrict__ src,
                                bf16_t* __restrict__ dst, int n, float scale) {
  int i = (blockIdx.x * blockDim.x + threadIdx.x) * 4;
  if (i >= n) return;
  const float4 f = *(const float4*)(src + i);
  v4bf o;
  o[0] = (__bf16)(f.x * scale);
  o[1] = (__bf16)(f.y * scale);
  o[2] = (__bf16)(f.z * scale);
  o[3] = (__bf16)(f.w * scale);
  *(v4bf*)(dst + i) = o;
}

// ---------------------------------------------------------------------------
// C[M,N] = A[M,K] @ W[N,K]^T, bf16 in, f32 acc.
// 128x128 block tile; TDM double-buffered LDS staging (padded rows: 40 elems);
// 8 waves in 4x2 grid, wave tile 32x64 = 2x4 WMMA accumulators.
// ---------------------------------------------------------------------------
template <typename OutT>
__global__ __launch_bounds__(256) void gemm_xwT(
    const bf16_t* __restrict__ A,   // [M][K]
    const bf16_t* __restrict__ W,   // [N][K]
    OutT* __restrict__ C,           // [M][N]
    int M, int N, int K) {
  __shared__ __align__(16) bf16_t At[2][128][40];  // 32 data + 8 pad per row
  __shared__ __align__(16) bf16_t Bt[2][128][40];

  const int lane = threadIdx.x & 31;
  const int wave = threadIdx.x >> 5;
  const int half = lane >> 4;
  const int l16  = lane & 15;
  const int wm = (wave & 3) * 32;          // wave M offset in tile
  const int wn = (wave >> 2) * 64;         // wave N offset in tile
  const int m0 = blockIdx.y * 128;
  const int n0 = blockIdx.x * 128;

  const v8f vzero = {0.f, 0.f, 0.f, 0.f, 0.f, 0.f, 0.f, 0.f};
  v8f acc[2][4];
#pragma unroll
  for (int mt = 0; mt < 2; ++mt)
#pragma unroll
    for (int nt = 0; nt < 4; ++nt) acc[mt][nt] = vzero;

  // Prologue: stage k=0 tiles into buffer 0.
  // Pad codes: interval 3 -> every 16 DWORDs (64B row), amount 3 -> 4 DWORDs (16B).
  if (wave == 0) {
    tdm_load_2d_bf16(lds_addr32(&At[0][0][0]), A + (size_t)m0 * K,
                     32, 128, (unsigned)K, (unsigned)M, (unsigned)K, 3, 3, true);
    tdm_load_2d_bf16(lds_addr32(&Bt[0][0][0]), W + (size_t)n0 * K,
                     32, 128, (unsigned)K, (unsigned)N, (unsigned)K, 3, 3, true);
  }

  int buf = 0;
  for (int k = 0; k < K; k += 32) {
    if (wave == 0) {
      if (k + 32 < K) {
        tdm_load_2d_bf16(lds_addr32(&At[buf ^ 1][0][0]),
                         A + (size_t)m0 * K + (k + 32),
                         32, 128, (unsigned)K, (unsigned)M, (unsigned)K, 3, 3, true);
        tdm_load_2d_bf16(lds_addr32(&Bt[buf ^ 1][0][0]),
                         W + (size_t)n0 * K + (k + 32),
                         32, 128, (unsigned)K, (unsigned)N, (unsigned)K, 3, 3, true);
        __builtin_amdgcn_s_wait_tensorcnt(2);  // current buf's 2 loads done
      } else {
        __builtin_amdgcn_s_wait_tensorcnt(0);
      }
    }
    __syncthreads();  // staged tiles visible to all waves

    v16bf af[2], bfg[4];
#pragma unroll
    for (int mt = 0; mt < 2; ++mt) {
      const bf16_t* p = &At[buf][wm + mt * 16 + l16][half * 8];
      af[mt] = cat8(*(const v8bf*)p, *(const v8bf*)(p + 16));
    }
#pragma unroll
    for (int nt = 0; nt < 4; ++nt) {
      const bf16_t* p = &Bt[buf][wn + nt * 16 + l16][half * 8];
      bfg[nt] = cat8(*(const v8bf*)p, *(const v8bf*)(p + 16));
    }
#pragma unroll
    for (int mt = 0; mt < 2; ++mt)
#pragma unroll
      for (int nt = 0; nt < 4; ++nt)
        acc[mt][nt] = __builtin_amdgcn_wmma_f32_16x16x32_bf16(
            false, af[mt], false, bfg[nt], (short)0, acc[mt][nt], false, false);

    __syncthreads();  // all reads done before this buffer is overwritten
    buf ^= 1;
  }

#pragma unroll
  for (int mt = 0; mt < 2; ++mt)
#pragma unroll
    for (int nt = 0; nt < 4; ++nt)
#pragma unroll
      for (int g = 0; g < 8; ++g) {
        const int row = m0 + wm + mt * 16 + g + 8 * half;
        const int col = n0 + wn + nt * 16 + l16;
        C[(size_t)row * N + col] = (OutT)acc[mt][nt][g];
      }
}

// ---------------------------------------------------------------------------
// Flash attention: block = (b, h, 128 q-rows); wave owns 16 q-rows.
// 64-key blocks: scores = 8 WMMA, PV = 8 WMMA per softmax pass.
// K tile staged by TDM (padded rows), V tile staged transposed by threads.
// ---------------------------------------------------------------------------
__global__ __launch_bounds__(256) void flash_attn(
    const bf16_t* __restrict__ Q,    // [B*S][D], pre-scaled by 1/sqrt(HD)
    const bf16_t* __restrict__ Km,   // [B*S][D]
    const bf16_t* __restrict__ Vm,   // [B*S][D]
    bf16_t* __restrict__ Ctx) {      // [B*S][D]
  __shared__ __align__(16) bf16_t Kt[64][72];      // K tile [key][hd], padded
  __shared__ __align__(16) bf16_t Vt[HD_][72];     // V^T tile [hd][key], padded
  __shared__ __align__(16) bf16_t Pb[8][16][64];   // per-wave P scratch

  const int lane = threadIdx.x & 31;
  const int wave = threadIdx.x >> 5;
  const int half = lane >> 4;
  const int l16  = lane & 15;

  const int bh   = blockIdx.x;
  const int b    = bh >> 4;
  const int h    = bh & 15;
  const int colh = h * HD_;
  const size_t rowBase = (size_t)b * S_;
  const int q0 = blockIdx.y * 128 + wave * 16;

  // Q fragments for this wave's 16 rows (K-dim = 64 -> two 16x32 fragments)
  v16bf qf[2];
  {
    const bf16_t* qp = Q + (rowBase + q0 + l16) * (size_t)D_ + colh;
#pragma unroll
    for (int kk = 0; kk < 2; ++kk) {
      const bf16_t* p = qp + kk * 32 + half * 8;
      qf[kk] = cat8(*(const v8bf*)p, *(const v8bf*)(p + 16));
    }
  }

  const v8f vzero = {0.f, 0.f, 0.f, 0.f, 0.f, 0.f, 0.f, 0.f};
  v8f acc[4];
#pragma unroll
  for (int nt = 0; nt < 4; ++nt) acc[nt] = vzero;
  float rowm[8], rowl[8];
#pragma unroll
  for (int g = 0; g < 8; ++g) { rowm[g] = -3.0e38f; rowl[g] = 0.f; }

  for (int kb = 0; kb < S_; kb += 64) {
    __syncthreads();  // prior iteration's Kt/Vt reads done
    if (wave == 0) {
      // 64(hd) x 64(key) tile; pad codes: interval 4 -> 32 DWORDs (128B row),
      // amount 3 -> 4 DWORDs (16B) => LDS row stride 144B = 72 elems.
      tdm_load_2d_bf16(lds_addr32(&Kt[0][0]),
                       Km + (rowBase + kb) * (size_t)D_ + colh,
                       64, 64, (unsigned)D_, (unsigned)(B_ * S_), (unsigned)D_,
                       4, 3, true);
    }
    {
      // transposed V staging: 4096 elems / 256 thr = 16 each
      const int key = threadIdx.x >> 2;
      const int hd0 = (threadIdx.x & 3) * 16;
      const bf16_t* vp = Vm + (rowBase + kb + key) * (size_t)D_ + colh + hd0;
      v8bf v0 = *(const v8bf*)vp;
      v8bf v1 = *(const v8bf*)(vp + 8);
#pragma unroll
      for (int i = 0; i < 8; ++i) Vt[hd0 + i][key] = v0[i];
#pragma unroll
      for (int i = 0; i < 8; ++i) Vt[hd0 + 8 + i][key] = v1[i];
    }
    if (wave == 0) __builtin_amdgcn_s_wait_tensorcnt(0);
    __syncthreads();  // Kt + Vt ready

    // ---- scores: 4 C tiles of 16 keys, K-dim 64 ----
    v8f sc[4];
#pragma unroll
    for (int ct = 0; ct < 4; ++ct) {
      v8f s = vzero;
      const bf16_t* kp = &Kt[ct * 16 + l16][0];
#pragma unroll
      for (int kk = 0; kk < 2; ++kk) {
        const bf16_t* p = kp + kk * 32 + half * 8;
        v16bf kf = cat8(*(const v8bf*)p, *(const v8bf*)(p + 16));
        s = __builtin_amdgcn_wmma_f32_16x16x32_bf16(false, qf[kk], false, kf,
                                                    (short)0, s, false, false);
      }
      sc[ct] = s;
    }

    // ---- online softmax (row = g + 8*half lives in one 16-lane half) ----
#pragma unroll
    for (int g = 0; g < 8; ++g) {
      float nm = fmaxf(fmaxf(sc[0][g], sc[1][g]), fmaxf(sc[2][g], sc[3][g]));
#pragma unroll
      for (int mask = 1; mask < 16; mask <<= 1)
        nm = fmaxf(nm, __shfl_xor(nm, mask, 32));
      const float newm = fmaxf(rowm[g], nm);
      const float corr = __expf(rowm[g] - newm);
      rowm[g] = newm;
      float rs = 0.f;
#pragma unroll
      for (int ct = 0; ct < 4; ++ct) {
        const float p = __expf(sc[ct][g] - newm);
        sc[ct][g] = p;
        rs += p;
      }
#pragma unroll
      for (int mask = 1; mask < 16; mask <<= 1)
        rs += __shfl_xor(rs, mask, 32);
      rowl[g] = rowl[g] * corr + rs;
#pragma unroll
      for (int nt = 0; nt < 4; ++nt) acc[nt][g] *= corr;
    }

    // ---- re-layout P: C layout -> A fragments via per-wave LDS ----
#pragma unroll
    for (int ct = 0; ct < 4; ++ct)
#pragma unroll
      for (int g = 0; g < 8; ++g)
        Pb[wave][g + 8 * half][ct * 16 + l16] = (__bf16)sc[ct][g];

    asm volatile("s_wait_dscnt 0" ::: "memory");

    const bf16_t* pp = &Pb[wave][l16][0];
    v16bf pf[2];
#pragma unroll
    for (int kk = 0; kk < 2; ++kk) {
      const bf16_t* p = pp + kk * 32 + half * 8;
      pf[kk] = cat8(*(const v8bf*)p, *(const v8bf*)(p + 16));
    }

    // ---- ctx += P (16x64) @ V_blk (64x64): 8 WMMAs from transposed LDS ----
#pragma unroll
    for (int nt = 0; nt < 4; ++nt)
#pragma unroll
      for (int kk = 0; kk < 2; ++kk) {
        const bf16_t* vp = &Vt[nt * 16 + l16][kk * 32 + half * 8];
        v16bf vf = cat8(*(const v8bf*)vp, *(const v8bf*)(vp + 16));
        acc[nt] = __builtin_amdgcn_wmma_f32_16x16x32_bf16(
            false, pf[kk], false, vf, (short)0, acc[nt], false, false);
      }
  }

  // epilogue: normalize by row sum, store bf16 ctx in [B*S][D] head layout
#pragma unroll
  for (int nt = 0; nt < 4; ++nt)
#pragma unroll
    for (int g = 0; g < 8; ++g) {
      const float o = acc[nt][g] / rowl[g];
      const size_t row = rowBase + q0 + g + 8 * half;
      Ctx[row * (size_t)D_ + colh + nt * 16 + l16] = (__bf16)o;
    }
}

// ---------------------------------------------------------------------------
extern "C" void kernel_launch(void* const* d_in, const int* in_sizes, int n_in,
                              void* d_out, int out_size, void* d_ws, size_t ws_size,
                              hipStream_t stream) {
  const float* x  = (const float*)d_in[0];
  const float* Wq = (const float*)d_in[1];
  const float* Wk = (const float*)d_in[2];
  const float* Wv = (const float*)d_in[3];
  const float* Wo = (const float*)d_in[4];
  float* out = (float*)d_out;

  char* ws = (char*)d_ws;
  const size_t MB = 1024 * 1024;
  bf16_t* xb  = (bf16_t*)(ws + 0 * MB);    // 16 MB  [B*S][D]
  bf16_t* wqb = (bf16_t*)(ws + 16 * MB);   //  2 MB  [D][D] (pre-scaled)
  bf16_t* wkb = (bf16_t*)(ws + 18 * MB);
  bf16_t* wvb = (bf16_t*)(ws + 20 * MB);
  bf16_t* wob = (bf16_t*)(ws + 22 * MB);
  bf16_t* Qb  = (bf16_t*)(ws + 24 * MB);   // 16 MB each
  bf16_t* Kb  = (bf16_t*)(ws + 40 * MB);
  bf16_t* Vb  = (bf16_t*)(ws + 56 * MB);
  bf16_t* Cb  = (bf16_t*)(ws + 72 * MB);   // attention context, bf16

  const int nX = B_ * S_ * D_;
  const int nW = D_ * D_;
  const float qscale = 0.125f;  // HD^-0.5 folded into Wq

  dim3 blk(256);
  cvt_f32_to_bf16<<<dim3((nX / 4 + 255) / 256), blk, 0, stream>>>(x,  xb,  nX, 1.0f);
  cvt_f32_to_bf16<<<dim3((nW / 4 + 255) / 256), blk, 0, stream>>>(Wq, wqb, nW, qscale);
  cvt_f32_to_bf16<<<dim3((nW / 4 + 255) / 256), blk, 0, stream>>>(Wk, wkb, nW, 1.0f);
  cvt_f32_to_bf16<<<dim3((nW / 4 + 255) / 256), blk, 0, stream>>>(Wv, wvb, nW, 1.0f);
  cvt_f32_to_bf16<<<dim3((nW / 4 + 255) / 256), blk, 0, stream>>>(Wo, wob, nW, 1.0f);

  dim3 ggrid(D_ / 128, (B_ * S_) / 128);  // (8, 64)
  gemm_xwT<bf16_t><<<ggrid, blk, 0, stream>>>(xb, wqb, Qb, B_ * S_, D_, D_);
  gemm_xwT<bf16_t><<<ggrid, blk, 0, stream>>>(xb, wkb, Kb, B_ * S_, D_, D_);
  gemm_xwT<bf16_t><<<ggrid, blk, 0, stream>>>(xb, wvb, Vb, B_ * S_, D_, D_);

  flash_attn<<<dim3(B_ * H_, S_ / 128), blk, 0, stream>>>(Qb, Kb, Vb, Cb);

  gemm_xwT<float><<<ggrid, blk, 0, stream>>>(Cb, wob, out, B_ * S_, D_, D_);
}
